// GlobalMamba_609885356385
// MI455X (gfx1250) — matrix operand order repaired and checked by reading, not compiled, not run on hardware
//
#include <hip/hip_runtime.h>
#include <stdint.h>

#define BATCH 4
#define N1T   131072
#define N2T   32768
#define DM    128
#define DI    256
#define DSN   16
#define QCH   256   // scan chunk length

typedef __attribute__((ext_vector_type(16))) __bf16          bf16x16;
typedef __attribute__((ext_vector_type(8)))  float           f32x8;
typedef __attribute__((ext_vector_type(8)))  unsigned short  u16x8;
typedef __attribute__((ext_vector_type(16))) unsigned short  u16x16;

__device__ __forceinline__ unsigned short f2bf(float f) {
  unsigned u = __builtin_bit_cast(unsigned, f);
  u += 0x7fffu + ((u >> 16) & 1u);            // round-to-nearest-even
  return (unsigned short)(u >> 16);
}

// ---------------------------------------------------------------- utilities
__global__ void k_convert_bf16(const float* __restrict__ src,
                               unsigned short* __restrict__ dst, int n) {
  int t = blockIdx.x * blockDim.x + threadIdx.x;
  if (t < n) dst[t] = f2bf(src[t]);
}

// f[row,d] = feats + pos@pe_w^T + pe_b
__global__ void k_pe(const float* __restrict__ feats, const float* __restrict__ pos,
                     const float* __restrict__ pw, const float* __restrict__ pb,
                     float* __restrict__ f, int total) {
  int t = blockIdx.x * blockDim.x + threadIdx.x;
  if (t >= total) return;
  int row = t >> 7, d = t & 127;
  float p0 = pos[row * 3 + 0], p1 = pos[row * 3 + 1], p2 = pos[row * 3 + 2];
  f[t] = feats[t] + p0 * pw[d * 3 + 0] + p1 * pw[d * 3 + 1] + p2 * pw[d * 3 + 2] + pb[d];
}

// ---------------------------------------------------------------- sort (bitonic, per batch)
__global__ void k_sort_init(const int* __restrict__ hil, long long* __restrict__ key,
                            int L, int total) {
  int t = blockIdx.x * blockDim.x + threadIdx.x;
  if (t >= total) return;
  key[t] = ((long long)hil[t] << 32) | (unsigned)(t % L);   // stable: tie-break by index
}

__global__ void k_bitonic(long long* __restrict__ key, int L, int j, int k) {
  int t = blockIdx.x * blockDim.x + threadIdx.x;   // over BATCH*L
  int i = t % L;
  int ixj = i ^ j;
  if (ixj <= i) return;
  long long* seg = key + (t - i);                  // batch segment base
  long long a = seg[i], b = seg[ixj];
  bool up = ((i & k) == 0);
  bool sw = up ? (a > b) : (a < b);
  if (sw) { seg[i] = b; seg[ixj] = a; }
}

// ---------------------------------------------------------------- gather + LayerNorm -> bf16
__global__ void k_gather_ln(const float* __restrict__ f, const long long* __restrict__ key,
                            const float* __restrict__ lg, const float* __restrict__ lb,
                            unsigned short* __restrict__ h, int L, int rev) {
  int rowseq = blockIdx.x * 8 + (threadIdx.x >> 5);   // one wave per sequence row
  int lane = threadIdx.x & 31;
  int bb = rowseq / L, l = rowseq - bb * L;
  int sl = rev ? (L - 1 - l) : l;
  int src = (int)(key[bb * L + sl] & 0xffffffffLL);
  const float* fr = f + (size_t)(bb * L + src) * DM;
  float v0 = fr[lane], v1 = fr[lane + 32], v2 = fr[lane + 64], v3 = fr[lane + 96];
  float s = v0 + v1 + v2 + v3;
#pragma unroll
  for (int o = 16; o; o >>= 1) s += __shfl_xor(s, o, 32);
  float mu = s * (1.0f / 128.0f);
  float d0 = v0 - mu, d1 = v1 - mu, d2 = v2 - mu, d3 = v3 - mu;
  float q = d0 * d0 + d1 * d1 + d2 * d2 + d3 * d3;
#pragma unroll
  for (int o = 16; o; o >>= 1) q += __shfl_xor(q, o, 32);
  float rs = rsqrtf(q * (1.0f / 128.0f) + 1e-5f);
  unsigned short* hr = h + (size_t)rowseq * DM;
  hr[lane]      = f2bf(d0 * rs * lg[lane]      + lb[lane]);
  hr[lane + 32] = f2bf(d1 * rs * lg[lane + 32] + lb[lane + 32]);
  hr[lane + 64] = f2bf(d2 * rs * lg[lane + 64] + lb[lane + 64]);
  hr[lane + 96] = f2bf(d3 * rs * lg[lane + 96] + lb[lane + 96]);
}

// ---------------------------------------------------------------- bf16 WMMA GEMM  C[M,N]=A[M,K]*W[N,K]^T
// B (weight) tile staged in LDS once per workgroup; A streamed from global.
__device__ __forceinline__ bf16x16 frag_a(const unsigned short* p, int half) {
  // A layout: lane(0-15) K = 0..15; lane(16-31) K = 16..31 (matches B)
  u16x8 a = *(const u16x8*)(p + half * 16);
  u16x8 b = *(const u16x8*)(p + half * 16 + 8);
  u16x16 c;
#pragma unroll
  for (int i = 0; i < 8; i++) { c[i] = a[i]; c[i + 8] = b[i]; }
  return __builtin_bit_cast(bf16x16, c);
}
__device__ __forceinline__ bf16x16 frag_b_lds(const unsigned short* p) {
  // lane(0-15) K=0..15, lane(16-31) K=16..31 ; two 16B ds loads (16B-aligned)
  u16x8 a = *(const u16x8*)p;
  u16x8 b = *(const u16x8*)(p + 8);
  u16x16 c;
#pragma unroll
  for (int i = 0; i < 8; i++) { c[i] = a[i]; c[i + 8] = b[i]; }
  return __builtin_bit_cast(bf16x16, c);
}
#define WMMA_BF16(a, b, c) \
  __builtin_amdgcn_wmma_f32_16x16x32_bf16(false, (a), false, (b), (short)0, (c), false, false)

__global__ void k_gemm_bf16(const unsigned short* __restrict__ A,
                            const unsigned short* __restrict__ W,
                            float* __restrict__ C, int M, int N, int K) {
  extern __shared__ unsigned short lw[];          // 64 x (K+8) halfs, 16B-aligned rows
  const int lstr = K + 8;                         // pad: 4-bank advance/row -> no conflicts
  int lane = threadIdx.x & 31;
  int wv = threadIdx.x >> 5;
  int m0 = (blockIdx.x * 8 + wv) * 16;            // grids are exact: m0 < M always
  int n0 = blockIdx.y * 64;

  // cooperative stage of W[n0:n0+64, 0:K] into LDS (u16x8 = 16B chunks)
  int chunks = K >> 3;
  for (int idx = threadIdx.x; idx < 64 * chunks; idx += blockDim.x) {
    int r = idx / chunks, kc = (idx - r * chunks) << 3;
    *(u16x8*)(lw + r * lstr + kc) = *(const u16x8*)(W + (size_t)(n0 + r) * K + kc);
  }
  __syncthreads();

  int half = lane >> 4, l16 = lane & 15;
  const unsigned short* arow = A + (size_t)(m0 + l16) * K;
  const unsigned short* b0 = lw + (l16 +  0) * lstr + half * 16;
  const unsigned short* b1 = lw + (l16 + 16) * lstr + half * 16;
  const unsigned short* b2 = lw + (l16 + 32) * lstr + half * 16;
  const unsigned short* b3 = lw + (l16 + 48) * lstr + half * 16;
  f32x8 c0 = {}, c1 = {}, c2 = {}, c3 = {};
  for (int k0 = 0; k0 < K; k0 += 32) {
    bf16x16 a = frag_a(arow + k0, half);
    c0 = WMMA_BF16(a, frag_b_lds(b0 + k0), c0);
    c1 = WMMA_BF16(a, frag_b_lds(b1 + k0), c1);
    c2 = WMMA_BF16(a, frag_b_lds(b2 + k0), c2);
    c3 = WMMA_BF16(a, frag_b_lds(b3 + k0), c3);
  }
  float* crow = C + (size_t)m0 * N + n0 + l16;
#pragma unroll
  for (int r = 0; r < 8; r++) {
    float* cr = crow + (size_t)(r + 8 * half) * N;
    cr[0] = c0[r]; cr[16] = c1[r]; cr[32] = c2[r]; cr[48] = c3[r];
  }
}

// ---------------------------------------------------------------- causal depthwise conv + SiLU
__global__ void k_conv_silu(const float* __restrict__ xz, const float* __restrict__ cw,
                            const float* __restrict__ cb, float* __restrict__ xc, int L, int total) {
  int t = blockIdx.x * blockDim.x + threadIdx.x;   // M*64
  if (t >= total) return;
  int row = t >> 6, d0 = (t & 63) << 2;
  int bb = row / L, l = row - bb * L;
  float a0 = cb[d0], a1 = cb[d0 + 1], a2 = cb[d0 + 2], a3 = cb[d0 + 3];
#pragma unroll
  for (int k = 0; k < 4; k++) {
    int ls = l - 3 + k;
    if (ls < 0) continue;
    float4 x = *(const float4*)(xz + (size_t)(bb * L + ls) * 512 + d0);
    a0 += cw[(d0 + 0) * 4 + k] * x.x;
    a1 += cw[(d0 + 1) * 4 + k] * x.y;
    a2 += cw[(d0 + 2) * 4 + k] * x.z;
    a3 += cw[(d0 + 3) * 4 + k] * x.w;
  }
  float* o = xc + (size_t)row * DI + d0;
  o[0] = a0 / (1.f + __expf(-a0));
  o[1] = a1 / (1.f + __expf(-a1));
  o[2] = a2 / (1.f + __expf(-a2));
  o[3] = a3 / (1.f + __expf(-a3));
}

// ---------------------------------------------------------------- x_proj (256 -> 40)
__global__ void k_xproj(const float* __restrict__ xc, const float* __restrict__ xw,
                        float* __restrict__ dbl) {
  int row = blockIdx.x;
  int j = threadIdx.x;
  if (j >= 40) return;
  const float* xr = xc + (size_t)row * DI;
  const float* wr = xw + j * DI;
  float acc = 0.f;
  for (int k = 0; k < DI; k++) acc += xr[k] * wr[k];
  dbl[(size_t)row * 40 + j] = acc;
}

// ---------------------------------------------------------------- dt = softplus(dbl[:, :8] @ dtw^T + dtb)
__global__ void k_dt(const float* __restrict__ dbl, const float* __restrict__ dtw,
                     const float* __restrict__ dtb, float* __restrict__ dt, int total) {
  int t = blockIdx.x * blockDim.x + threadIdx.x;   // M*256
  if (t >= total) return;
  int row = t >> 8, d = t & 255;
  const float* dr = dbl + (size_t)row * 40;
  const float* wr = dtw + d * 8;
  float acc = dtb[d];
#pragma unroll
  for (int r = 0; r < 8; r++) acc += dr[r] * wr[r];
  dt[t] = (acc > 20.f) ? acc : log1pf(__expf(acc));
}

// ---------------------------------------------------------------- chunked selective scan
// pass1: per (b,d,chunk,s): P = prod a_t, S = local state with h0=0
__global__ void k_scan1(const float* __restrict__ dt, const float* __restrict__ u,
                        const float* __restrict__ dbl, const float* __restrict__ A_log,
                        float* __restrict__ cP, float* __restrict__ cS, int L, int C) {
  int t = blockIdx.x * blockDim.x + threadIdx.x;   // B*256*C*16
  int s = t & 15;
  int g = t >> 4;
  int c = g % C;
  int tmp = g / C;
  int d = tmp & 255, bb = tmp >> 8;
  float A = -__expf(A_log[d * 16 + s]);
  float P = 1.f, S = 0.f;
  int base = bb * L + c * QCH;
  for (int q = 0; q < QCH; q++) {
    int row = base + q;
    float dtv = dt[(size_t)row * 256 + d];
    float uv  = u[(size_t)row * 256 + d];
    float Bv  = dbl[(size_t)row * 40 + 8 + s];
    float a = __expf(dtv * A);
    S = a * S + dtv * uv * Bv;
    P *= a;
  }
  size_t o = ((size_t)(bb * 256 + d) * C + c) * 16 + s;
  cP[o] = P; cS[o] = S;
}

// pass2: sequential combine over chunks; overwrites cS slots with chunk initial states
__global__ void k_scan2(const float* __restrict__ cP, float* __restrict__ cS, int C) {
  int t = blockIdx.x * blockDim.x + threadIdx.x;   // B*256*16
  int s = t & 15, d = (t >> 4) & 255, bb = t >> 12;
  float H = 0.f;
  size_t baseo = (size_t)(bb * 256 + d) * C * 16 + s;
  for (int c = 0; c < C; c++) {
    size_t o = baseo + (size_t)c * 16;
    float P = cP[o], S = cS[o];
    cS[o] = H;                                   // Hstart for this chunk
    H = P * H + S;
  }
}

// pass3: replay chunk with correct init; fuse y+u*D, SiLU(z) gate, bf16 convert
__global__ void k_scan3(const float* __restrict__ dt, const float* __restrict__ u,
                        const float* __restrict__ dbl, const float* __restrict__ A_log,
                        const float* __restrict__ Hs, const float* __restrict__ xz,
                        const float* __restrict__ Dp, unsigned short* __restrict__ gout,
                        int L, int C) {
  int t = blockIdx.x * blockDim.x + threadIdx.x;   // B*256*C
  int c = t % C;
  int tmp = t / C;
  int d = tmp & 255, bb = tmp >> 8;
  float Aa[16], hh[16];
  size_t ho = ((size_t)(bb * 256 + d) * C + c) * 16;
#pragma unroll
  for (int s = 0; s < 16; s++) {
    Aa[s] = -__expf(A_log[d * 16 + s]);
    hh[s] = Hs[ho + s];
  }
  float Dd = Dp[d];
  int base = bb * L + c * QCH;
  for (int q = 0; q < QCH; q++) {
    int row = base + q;
    float dtv = dt[(size_t)row * 256 + d];
    float uv  = u[(size_t)row * 256 + d];
    const float* drow = dbl + (size_t)row * 40;
    float du = dtv * uv, y = 0.f;
#define UPD(S, BV, CV)                                   \
    { float a_ = __expf(dtv * Aa[S]);                    \
      hh[S] = a_ * hh[S] + du * (BV);                    \
      y += hh[S] * (CV); }
#pragma unroll
    for (int j = 0; j < 4; j++) {
      float4 B4 = ((const float4*)(drow + 8))[j];
      float4 C4 = ((const float4*)(drow + 24))[j];
      UPD(4 * j + 0, B4.x, C4.x);
      UPD(4 * j + 1, B4.y, C4.y);
      UPD(4 * j + 2, B4.z, C4.z);
      UPD(4 * j + 3, B4.w, C4.w);
    }
#undef UPD
    y += uv * Dd;
    float zv = xz[(size_t)row * 512 + 256 + d];
    float sz = zv / (1.f + __expf(-zv));
    gout[(size_t)row * 256 + d] = f2bf(y * sz);
  }
}

// ---------------------------------------------------------------- scatter-back + final LayerNorm
__global__ void k_scatter_ln(const float* __restrict__ ymm, const long long* __restrict__ key,
                             const float* __restrict__ lg, const float* __restrict__ lb,
                             float* __restrict__ out, int L, int rev) {
  int rowseq = blockIdx.x * 8 + (threadIdx.x >> 5);
  int lane = threadIdx.x & 31;
  int bb = rowseq / L, l = rowseq - bb * L;
  int sl = rev ? (L - 1 - l) : l;
  int dst = (int)(key[bb * L + sl] & 0xffffffffLL);
  const float* yr = ymm + (size_t)rowseq * DM;
  float v0 = yr[lane], v1 = yr[lane + 32], v2 = yr[lane + 64], v3 = yr[lane + 96];
  float s = v0 + v1 + v2 + v3;
#pragma unroll
  for (int o = 16; o; o >>= 1) s += __shfl_xor(s, o, 32);
  float mu = s * (1.0f / 128.0f);
  float d0 = v0 - mu, d1 = v1 - mu, d2 = v2 - mu, d3 = v3 - mu;
  float q = d0 * d0 + d1 * d1 + d2 * d2 + d3 * d3;
#pragma unroll
  for (int o = 16; o; o >>= 1) q += __shfl_xor(q, o, 32);
  float rs = rsqrtf(q * (1.0f / 128.0f) + 1e-5f);
  float* orow = out + (size_t)(bb * L + dst) * DM;
  orow[lane]      = d0 * rs * lg[lane]      + lb[lane];
  orow[lane + 32] = d1 * rs * lg[lane + 32] + lb[lane + 32];
  orow[lane + 64] = d2 * rs * lg[lane + 64] + lb[lane + 64];
  orow[lane + 96] = d3 * rs * lg[lane + 96] + lb[lane + 96];
}

// ================================================================ host side
struct WS {
  long long* key;
  float *f, *xz, *xc, *dbl, *dt, *ymm, *cP, *cS;
  unsigned short *h, *g, *win, *wout;
};

static void run_stream(const float* feats, const float* pos, const int* hil, int Nn,
                       const float* const* mp,          // 11 mamba params
                       const float* pe_w, const float* pe_b,
                       const float* fg, const float* fb, int rev,
                       float* outp, const WS& w, hipStream_t stream) {
  const int L = Nn / BATCH;
  const int C = L / QCH;
  const float *ln_g = mp[0], *ln_b = mp[1], *in_w = mp[2], *conv_w = mp[3], *conv_b = mp[4],
              *xproj_w = mp[5], *dtw = mp[6], *dtb = mp[7], *A_log = mp[8], *Dp = mp[9],
              *out_w = mp[10];

  k_convert_bf16<<<(512 * 128) / 256, 256, 0, stream>>>(in_w, w.win, 512 * 128);
  k_convert_bf16<<<(128 * 256) / 256, 256, 0, stream>>>(out_w, w.wout, 128 * 256);

  k_pe<<<(Nn * 128) / 256, 256, 0, stream>>>(feats, pos, pe_w, pe_b, w.f, Nn * 128);

  k_sort_init<<<Nn / 256, 256, 0, stream>>>(hil, w.key, L, Nn);
  for (int k = 2; k <= L; k <<= 1)
    for (int j = k >> 1; j > 0; j >>= 1)
      k_bitonic<<<Nn / 256, 256, 0, stream>>>(w.key, L, j, k);

  k_gather_ln<<<Nn / 8, 256, 0, stream>>>(w.f, w.key, ln_g, ln_b, w.h, L, rev);

  { // in_proj: [Nn,128] x [512,128]^T -> [Nn,512]
    dim3 g(Nn / 128, 512 / 64);
    size_t shm = 64 * (128 + 8) * sizeof(unsigned short);
    k_gemm_bf16<<<g, 256, shm, stream>>>(w.h, w.win, w.xz, Nn, 512, 128);
  }

  k_conv_silu<<<(Nn * 64) / 256, 256, 0, stream>>>(w.xz, conv_w, conv_b, w.xc, L, Nn * 64);
  k_xproj<<<Nn, 64, 0, stream>>>(w.xc, xproj_w, w.dbl);
  k_dt<<<Nn, 256, 0, stream>>>(w.dbl, dtw, dtb, w.dt, Nn * 256);

  k_scan1<<<Nn / 16, 256, 0, stream>>>(w.dt, w.xc, w.dbl, A_log, w.cP, w.cS, L, C);
  k_scan2<<<(BATCH * 256 * 16) / 256, 256, 0, stream>>>(w.cP, w.cS, C);
  k_scan3<<<Nn / 256, 256, 0, stream>>>(w.dt, w.xc, w.dbl, A_log, w.cS, w.xz, Dp, w.g, L, C);

  { // out_proj: [Nn,256] x [128,256]^T -> [Nn,128]
    dim3 g(Nn / 128, 128 / 64);
    size_t shm = 64 * (256 + 8) * sizeof(unsigned short);
    k_gemm_bf16<<<g, 256, shm, stream>>>(w.g, w.wout, w.ymm, Nn, 128, 256);
  }

  k_scatter_ln<<<Nn / 8, 256, 0, stream>>>(w.ymm, w.key, fg, fb, outp, L, rev);
}

extern "C" void kernel_launch(void* const* d_in, const int* in_sizes, int n_in,
                              void* d_out, int out_size, void* d_ws, size_t ws_size,
                              hipStream_t stream) {
  (void)in_sizes; (void)n_in; (void)out_size; (void)ws_size;
  const float* feats_s1 = (const float*)d_in[0];
  const float* feats_s2 = (const float*)d_in[1];
  const float* pos_s1   = (const float*)d_in[2];
  const float* pos_s2   = (const float*)d_in[3];
  const int*   hil_s1   = (const int*)d_in[4];
  const int*   hil_s2   = (const int*)d_in[5];
  const float* pe_w     = (const float*)d_in[6];
  const float* pe_b     = (const float*)d_in[7];
  const float* norm_g   = (const float*)d_in[8];
  const float* norm_b   = (const float*)d_in[9];
  const float* normb_g  = (const float*)d_in[10];
  const float* normb_b  = (const float*)d_in[11];
  const float* m1[11];
  const float* m2[11];
  for (int i = 0; i < 11; i++) { m1[i] = (const float*)d_in[12 + i]; m2[i] = (const float*)d_in[23 + i]; }

  // workspace layout (sized for the larger stream; streams run sequentially)
  char* ws = (char*)d_ws;
  size_t off = 0;
  auto alloc = [&](size_t bytes) { size_t r = off; off += (bytes + 255) & ~(size_t)255; return r; };
  const size_t NMAX = N1T;
  const size_t CMAX = (NMAX / BATCH) / QCH;
  WS w;
  w.key  = (long long*)(ws + alloc(NMAX * 8));
  w.f    = (float*)(ws + alloc(NMAX * DM * 4));
  w.h    = (unsigned short*)(ws + alloc(NMAX * DM * 2));
  w.xz   = (float*)(ws + alloc(NMAX * 512 * 4));
  w.xc   = (float*)(ws + alloc(NMAX * DI * 4));
  w.dbl  = (float*)(ws + alloc(NMAX * 40 * 4));
  w.dt   = (float*)(ws + alloc(NMAX * DI * 4));
  w.g    = (unsigned short*)(ws + alloc(NMAX * DI * 2));
  w.ymm  = (float*)(ws + alloc(NMAX * DM * 4));
  w.cP   = (float*)(ws + alloc((size_t)BATCH * 256 * CMAX * 16 * 4));
  w.cS   = (float*)(ws + alloc((size_t)BATCH * 256 * CMAX * 16 * 4));
  w.win  = (unsigned short*)(ws + alloc(512 * 128 * 2));
  w.wout = (unsigned short*)(ws + alloc(128 * 256 * 2));

  float* out = (float*)d_out;
  // s2: mamba block m1, forward order, final LN (norm_g/b), rows [N1, N1+N2)
  run_stream(feats_s2, pos_s2, hil_s2, N2T, m1, pe_w, pe_b,
             norm_g, norm_b, /*rev=*/0, out + (size_t)N1T * DM, w, stream);
  // s1: mamba block m2, reversed order, final LN (normb_g/b), rows [0, N1)
  run_stream(feats_s1, pos_s1, hil_s1, N1T, m2, pe_w, pe_b,
             normb_g, normb_b, /*rev=*/1, out, w, stream);
}
